// Cfconv_46746424050014
// MI455X (gfx1250) — matrix operand order repaired and checked by reading, not compiled, not run on hardware
//
#include <hip/hip_runtime.h>
#include <hip/hip_bf16.h>

// ---------------- problem constants ----------------
constexpr int NAT   = 50000;   // atoms
constexpr int NEDGE = 800000;  // edges
constexpr int NF    = 128;     // features
constexpr int ND    = 50;      // rbf features (padded to 64 for K)
constexpr float RCUT    = 5.0f;
constexpr float SPACING = RCUT / (ND - 1);
constexpr float RBFCOEF = -0.5f / (SPACING * SPACING);

// ---------------- WMMA types ----------------
typedef __attribute__((ext_vector_type(16))) __bf16 v16bf;
typedef __attribute__((ext_vector_type(8)))  __bf16 v8bf;
typedef __attribute__((ext_vector_type(8)))  float  v8f;

static __device__ __forceinline__ v8f wmma_bf16(v16bf a, v16bf b, v8f c) {
    // D = A(16x32 bf16) x B(32x16 bf16) + C(16x16 f32)
    return __builtin_amdgcn_wmma_f32_16x16x32_bf16(
        false, a, false, b, (short)0, c, false, false);
}

static __device__ __forceinline__ float sspf(float x) {
    float sp = (x > 15.f) ? x : __logf(1.f + __expf(x));
    return sp - 0.69314718056f;
}

// ---------------- LDS B-matrix staging ----------------
// B layout for 16-bit 32x16 tile (wave32): lane 0-15 -> N=lane, K={0..7,16..23};
// lane 16-31 -> N=lane-16, K={8..15,24..31}.  We pre-swizzle weights (row-major KxN
// f32 in global) into per-(ktile,ntile) 1KB blocks: block*1024B + lane*32B.
static __device__ void stage_B(const float* __restrict__ W, __bf16* lds,
                               int Kact, int nKT, int tid, int nth) {
    int total = nKT * 8 * 512;
    for (int idx = tid; idx < total; idx += nth) {
        int b    = idx >> 9;          // (kt*8 + nt)
        int rem  = idx & 511;
        int lane = rem >> 4;
        int j    = rem & 15;
        int kt   = b >> 3;
        int nt   = b & 7;
        int k = kt * 32 + ((lane < 16) ? 0 : 8) + j + ((j < 8) ? 0 : 8);
        int n = nt * 16 + (lane & 15);
        float v = (k < Kact) ? W[k * NF + n] : 0.f;
        lds[idx] = (__bf16)v;
    }
}

static __device__ __forceinline__ v16bf load_b(const __bf16* lds, int kt, int nt, int lane) {
    return *(const v16bf*)(lds + (((kt << 3) + nt) << 9) + (lane << 4));
}

// A layout for 16-bit 16x32 tile: lane 0-15: row M=lane, K={0..7,16..23};
// lane 16-31: row M=lane-15? -> M=lane&15, K={8..15,24..31}. kb = 0 or 8.
static __device__ __forceinline__ v16bf load_a_f32(const float* __restrict__ rowp,
                                                   int kt, int kb) {
    const float* p = rowp + kt * 32 + kb;
    v16bf a;
#pragma unroll
    for (int i = 0; i < 8; ++i) {
        a[i]     = (__bf16)p[i];
        a[8 + i] = (__bf16)p[16 + i];
    }
    return a;
}

static __device__ __forceinline__ v16bf load_a_h(const __bf16* h, int M, int kt, int kb) {
    const __bf16* p = h + M * NF + kt * 32 + kb;
    v8bf lo = *(const v8bf*)p;
    v8bf hi = *(const v8bf*)(p + 16);
    v16bf a;
#pragma unroll
    for (int i = 0; i < 8; ++i) { a[i] = lo[i]; a[8 + i] = hi[i]; }
    return a;
}

// ---------------- small kernels ----------------
__global__ void init_x_kernel(const float* __restrict__ emb, const int* __restrict__ Z,
                              float* __restrict__ X) {
    int total = NAT * NF;
    for (int i = blockIdx.x * blockDim.x + threadIdx.x; i < total;
         i += gridDim.x * blockDim.x) {
        int n = i >> 7, f = i & 127;
        X[i] = emb[Z[n] * NF + f];
    }
}

__global__ void zero_kernel(float* __restrict__ p, int n) {
    for (int i = blockIdx.x * blockDim.x + threadIdx.x; i < n;
         i += gridDim.x * blockDim.x) p[i] = 0.f;
}

__global__ void edge_geom_kernel(const float* __restrict__ R,
                                 const int* __restrict__ idxI, const int* __restrict__ idxJ,
                                 float* __restrict__ dE, float* __restrict__ rcE) {
    for (int e = blockIdx.x * blockDim.x + threadIdx.x; e < NEDGE;
         e += gridDim.x * blockDim.x) {
        int i = idxI[e], j = idxJ[e];
        float dx = R[3 * j + 0] - R[3 * i + 0];
        float dy = R[3 * j + 1] - R[3 * i + 1];
        float dz = R[3 * j + 2] - R[3 * i + 2];
        float d = sqrtf(dx * dx + dy * dy + dz * dz);
        float rc = (d < RCUT) ? 0.5f * (__cosf(d * (3.14159265358979f / RCUT)) + 1.f) : 0.f;
        dE[e] = d;
        rcE[e] = rc;
    }
}

// ---------------- node pre: xf(bf16) = X @ W  ----------------
__global__ void __launch_bounds__(256)
node_pre_kernel(const float* __restrict__ X, const float* __restrict__ W,
                __bf16* __restrict__ xf) {
    __shared__ __align__(32) __bf16 sW[32 * 512];
    const int tid = threadIdx.x, lane = tid & 31;
    stage_B(W, sW, NF, 4, tid, blockDim.x);
    __syncthreads();

    const int M = lane & 15;
    const int kb = (lane < 16) ? 0 : 8;
    const int half8 = (lane < 16) ? 0 : 8;
    const int colL = lane & 15;
    const int nwaves = (gridDim.x * blockDim.x) >> 5;
    int gw = (blockIdx.x * blockDim.x + tid) >> 5;

    for (int tile = gw; tile < NAT / 16; tile += nwaves) {
        v8f acc[8];
#pragma unroll
        for (int nt = 0; nt < 8; ++nt)
#pragma unroll
            for (int r = 0; r < 8; ++r) acc[nt][r] = 0.f;
#pragma unroll
        for (int kt = 0; kt < 4; ++kt) {
            v16bf a = load_a_f32(X + (tile * 16 + M) * NF, kt, kb);
#pragma unroll
            for (int nt = 0; nt < 8; ++nt)
                acc[nt] = wmma_bf16(a, load_b(sW, kt, nt, lane), acc[nt]);
        }
#pragma unroll
        for (int nt = 0; nt < 8; ++nt) {
            int col = nt * 16 + colL;
#pragma unroll
            for (int r = 0; r < 8; ++r)
                xf[(tile * 16 + r + half8) * NF + col] = (__bf16)acc[nt][r];
        }
    }
}

// ---------------- fused edge filter kernel ----------------
__global__ void __launch_bounds__(256)
edge_filter_kernel(const float* __restrict__ dE, const float* __restrict__ rcE,
                   const int* __restrict__ idxI, const int* __restrict__ idxJ,
                   const __bf16* __restrict__ xf, float* __restrict__ agg,
                   const float* __restrict__ W1, const float* __restrict__ b1,
                   const float* __restrict__ W2, const float* __restrict__ b2) {
    __shared__ __align__(32) __bf16 sW1[16 * 512];      // 64x128 (K padded)
    __shared__ __align__(32) __bf16 sW2[32 * 512];      // 128x128
    __shared__ float sb1[NF], sb2[NF];
    __shared__ __align__(32) __bf16 sH[8][16 * NF];     // per-wave hidden tile
    __shared__ float sD[8][16], sRc[8][16];
    __shared__ int   sI[8][16], sJ[8][16];

    const int tid = threadIdx.x, lane = tid & 31, w = tid >> 5;
    stage_B(W1, sW1, ND, 2, tid, blockDim.x);
    stage_B(W2, sW2, NF, 4, tid, blockDim.x);
    for (int f = tid; f < NF; f += blockDim.x) { sb1[f] = b1[f]; sb2[f] = b2[f]; }
    __syncthreads();

    const int M = lane & 15;
    const int kb = (lane < 16) ? 0 : 8;
    const int half8 = (lane < 16) ? 0 : 8;
    const int colL = lane & 15;
    const int nwaves = (gridDim.x * blockDim.x) >> 5;
    int gw = (blockIdx.x * blockDim.x + tid) >> 5;
    const int ntiles = NEDGE / 16;

    for (int tile = gw; tile < ntiles; tile += nwaves) {
        int eb = tile * 16;
        if (lane < 16) {
            sD[w][lane]  = dE[eb + lane];
            sRc[w][lane] = rcE[eb + lane];
            sI[w][lane]  = idxI[eb + lane];
            sJ[w][lane]  = idxJ[eb + lane];
        }
        float dd = sD[w][M];

        // GEMM1: rbf(d) [16x64] @ W1 [64x128], bias b1
        v8f acc[8];
#pragma unroll
        for (int nt = 0; nt < 8; ++nt) {
            float bv = sb1[nt * 16 + colL];
#pragma unroll
            for (int r = 0; r < 8; ++r) acc[nt][r] = bv;
        }
#pragma unroll
        for (int kt = 0; kt < 2; ++kt) {
            v16bf a;
#pragma unroll
            for (int j = 0; j < 8; ++j) {
                int k0 = kt * 32 + kb + j;
                int k1 = k0 + 16;
                float t0 = dd - (float)k0 * SPACING;
                float t1 = dd - (float)k1 * SPACING;
                float v0 = (k0 < ND) ? __expf(RBFCOEF * t0 * t0) : 0.f;
                float v1 = (k1 < ND) ? __expf(RBFCOEF * t1 * t1) : 0.f;
                a[j]     = (__bf16)v0;
                a[8 + j] = (__bf16)v1;
            }
#pragma unroll
            for (int nt = 0; nt < 8; ++nt)
                acc[nt] = wmma_bf16(a, load_b(sW1, kt, nt, lane), acc[nt]);
        }

        // ssp -> per-wave LDS hidden tile (bf16)
        __bf16* hrow = &sH[w][0];
#pragma unroll
        for (int nt = 0; nt < 8; ++nt) {
            int col = nt * 16 + colL;
#pragma unroll
            for (int r = 0; r < 8; ++r)
                hrow[(r + half8) * NF + col] = (__bf16)sspf(acc[nt][r]);
        }

        // GEMM2: h [16x128] @ W2 [128x128], bias b2
        v8f acc2[8];
#pragma unroll
        for (int nt = 0; nt < 8; ++nt) {
            float bv = sb2[nt * 16 + colL];
#pragma unroll
            for (int r = 0; r < 8; ++r) acc2[nt][r] = bv;
        }
#pragma unroll
        for (int kt = 0; kt < 4; ++kt) {
            v16bf a = load_a_h(hrow, M, kt, kb);
#pragma unroll
            for (int nt = 0; nt < 8; ++nt)
                acc2[nt] = wmma_bf16(a, load_b(sW2, kt, nt, lane), acc2[nt]);
        }

        // Wij * rcut * xf[idx_j]  -> atomic scatter into agg[idx_i]
#pragma unroll
        for (int nt = 0; nt < 8; ++nt) {
            int col = nt * 16 + colL;
#pragma unroll
            for (int r = 0; r < 8; ++r) {
                int rowt = r + half8;
                float wij = acc2[nt][r] * sRc[w][rowt];
                float xv  = (float)xf[sJ[w][rowt] * NF + col];
                atomicAdd(&agg[sI[w][rowt] * NF + col], wij * xv);
            }
        }
    }
}

// ---------------- node post: X += ssp(agg@W1+b1)@W2 + b2 ----------------
__global__ void __launch_bounds__(256)
node_post_kernel(float* __restrict__ X, const float* __restrict__ agg,
                 const float* __restrict__ W1, const float* __restrict__ b1,
                 const float* __restrict__ W2, const float* __restrict__ b2) {
    __shared__ __align__(32) __bf16 sW1[32 * 512];
    __shared__ __align__(32) __bf16 sW2[32 * 512];
    __shared__ float sb1[NF], sb2[NF];
    __shared__ __align__(32) __bf16 sH[8][16 * NF];

    const int tid = threadIdx.x, lane = tid & 31, w = tid >> 5;
    stage_B(W1, sW1, NF, 4, tid, blockDim.x);
    stage_B(W2, sW2, NF, 4, tid, blockDim.x);
    for (int f = tid; f < NF; f += blockDim.x) { sb1[f] = b1[f]; sb2[f] = b2[f]; }
    __syncthreads();

    const int M = lane & 15;
    const int kb = (lane < 16) ? 0 : 8;
    const int half8 = (lane < 16) ? 0 : 8;
    const int colL = lane & 15;
    const int nwaves = (gridDim.x * blockDim.x) >> 5;
    int gw = (blockIdx.x * blockDim.x + tid) >> 5;

    for (int tile = gw; tile < NAT / 16; tile += nwaves) {
        v8f acc[8];
#pragma unroll
        for (int nt = 0; nt < 8; ++nt) {
            float bv = sb1[nt * 16 + colL];
#pragma unroll
            for (int r = 0; r < 8; ++r) acc[nt][r] = bv;
        }
#pragma unroll
        for (int kt = 0; kt < 4; ++kt) {
            v16bf a = load_a_f32(agg + (tile * 16 + M) * NF, kt, kb);
#pragma unroll
            for (int nt = 0; nt < 8; ++nt)
                acc[nt] = wmma_bf16(a, load_b(sW1, kt, nt, lane), acc[nt]);
        }
        __bf16* hrow = &sH[w][0];
#pragma unroll
        for (int nt = 0; nt < 8; ++nt) {
            int col = nt * 16 + colL;
#pragma unroll
            for (int r = 0; r < 8; ++r)
                hrow[(r + half8) * NF + col] = (__bf16)sspf(acc[nt][r]);
        }
        v8f acc2[8];
#pragma unroll
        for (int nt = 0; nt < 8; ++nt) {
            float bv = sb2[nt * 16 + colL];
#pragma unroll
            for (int r = 0; r < 8; ++r) acc2[nt][r] = bv;
        }
#pragma unroll
        for (int kt = 0; kt < 4; ++kt) {
            v16bf a = load_a_h(hrow, M, kt, kb);
#pragma unroll
            for (int nt = 0; nt < 8; ++nt)
                acc2[nt] = wmma_bf16(a, load_b(sW2, kt, nt, lane), acc2[nt]);
        }
#pragma unroll
        for (int nt = 0; nt < 8; ++nt) {
            int col = nt * 16 + colL;
#pragma unroll
            for (int r = 0; r < 8; ++r) {
                int idx = (tile * 16 + r + half8) * NF + col;
                X[idx] = X[idx] + acc2[nt][r];
            }
        }
    }
}

// ---------------- launch ----------------
extern "C" void kernel_launch(void* const* d_in, const int* in_sizes, int n_in,
                              void* d_out, int out_size, void* d_ws, size_t ws_size,
                              hipStream_t stream) {
    const float* R      = (const float*)d_in[0];
    const int*   Z      = (const int*)d_in[1];
    const int*   idx_i  = (const int*)d_in[2];
    const int*   idx_j  = (const int*)d_in[3];
    const float* emb    = (const float*)d_in[4];
    const float* in2f_W = (const float*)d_in[5];
    const float* fn_W1  = (const float*)d_in[6];
    const float* fn_b1  = (const float*)d_in[7];
    const float* fn_W2  = (const float*)d_in[8];
    const float* fn_b2  = (const float*)d_in[9];
    const float* o_W1   = (const float*)d_in[10];
    const float* o_b1   = (const float*)d_in[11];
    const float* o_W2   = (const float*)d_in[12];
    const float* o_b2   = (const float*)d_in[13];

    float* X  = (float*)d_out;
    char*  ws = (char*)d_ws;
    float*  dE  = (float*)(ws + 0);                       // E * 4   = 3.2 MB
    float*  rcE = (float*)(ws + 3200000);                 // E * 4   = 3.2 MB
    __bf16* xf  = (__bf16*)(ws + 6400000);                // N*F * 2 = 12.8 MB
    float*  agg = (float*)(ws + 19200000);                // N*F * 4 = 25.6 MB

    init_x_kernel<<<2048, 256, 0, stream>>>(emb, Z, X);
    edge_geom_kernel<<<1024, 256, 0, stream>>>(R, idx_i, idx_j, dE, rcE);

    for (int l = 0; l < 3; ++l) {
        node_pre_kernel<<<256, 256, 0, stream>>>(X, in2f_W + (size_t)l * NF * NF, xf);
        zero_kernel<<<2048, 256, 0, stream>>>(agg, NAT * NF);
        edge_filter_kernel<<<512, 256, 0, stream>>>(
            dE, rcE, idx_i, idx_j, xf, agg,
            fn_W1 + (size_t)l * ND * NF, fn_b1 + (size_t)l * NF,
            fn_W2 + (size_t)l * NF * NF, fn_b2 + (size_t)l * NF);
        node_post_kernel<<<256, 256, 0, stream>>>(
            X, agg,
            o_W1 + (size_t)l * NF * NF, o_b1 + (size_t)l * NF,
            o_W2 + (size_t)l * NF * NF, o_b2 + (size_t)l * NF);
    }
}
